// DownConv_40544491274261
// MI455X (gfx1250) — matrix operand order repaired and checked by reading, not compiled, not run on hardware
//
#include <hip/hip_runtime.h>

// ---------------------------------------------------------------------------
// Types / WMMA helpers (CDNA5 wave32, V_WMMA_F32_16X16X32_F16)
// ---------------------------------------------------------------------------
typedef __attribute__((ext_vector_type(16))) _Float16 v16h;
typedef __attribute__((ext_vector_type(8)))  float    v8f;
typedef __attribute__((ext_vector_type(4)))  float    f32x4;

__device__ __forceinline__ v8f wmma16(v16h a, v16h b, v8f c) {
    return __builtin_amdgcn_wmma_f32_16x16x32_f16(false, a, false, b, (short)0, c,
                                                  false, false);
}

// A-matrix fragment (16x32 f16), row-major source with leading dim `ld` halves.
// Lane l: m=l&15, h=l>>4. elems 0..7 <- K=8h..8h+7 ; elems 8..15 <- K=16+8h..
__device__ __forceinline__ v16h frag_a(const _Float16* base, int ld) {
    int l = threadIdx.x & 31;
    int m = l & 15, h = l >> 4;
    const _Float16* p = base + (size_t)m * ld + 8 * h;
    v16h r;
    ((f32x4*)&r)[0] = *(const f32x4*)(p);
    ((f32x4*)&r)[1] = *(const f32x4*)(p + 16);
    return r;
}

// B-matrix fragment from B^T layout [n][k] with leading dim `ld` halves.
// Lane l: n=l&15, h=l>>4. elems 0..7 <- K=8h..8h+7 ; elems 8..15 <- K=16+8h..
__device__ __forceinline__ v16h frag_b(const _Float16* base, int ld) {
    int l = threadIdx.x & 31;
    int n = l & 15, h = l >> 4;
    const _Float16* p = base + (size_t)n * ld + 8 * h;
    v16h r;
    ((f32x4*)&r)[0] = *(const f32x4*)(p);
    ((f32x4*)&r)[1] = *(const f32x4*)(p + 16);
    return r;
}

// ---------------------------------------------------------------------------
// Problem constants
// ---------------------------------------------------------------------------
#define BB    2
#define NP    100000
#define NPTS  200000        // B*N
#define RR    64
#define RRR   262144        // 64^3
#define BRRR  524288
#define CINC  32
#define COUTC 64
#define WSZ   125
#define INV_NORM 0.9082652134423251f   // 1/(1+0.1+0.001)

// ---------------------------------------------------------------------------
// Small prep kernels
// ---------------------------------------------------------------------------
__global__ void k_zero_f32(float* p, size_t n) {
    size_t i = (size_t)blockIdx.x * blockDim.x + threadIdx.x;
    size_t st = (size_t)gridDim.x * blockDim.x;
    for (; i < n; i += st) p[i] = 0.f;
}

__global__ void k_cvt16(const float* __restrict__ s, _Float16* __restrict__ d, int n) {
    int i = blockIdx.x * blockDim.x + threadIdx.x;
    if (i < n) d[i] = (_Float16)s[i];
}

// B^T prep: W is [K][N] f32 -> out [N][K] f16
__global__ void k_wprep_t(const float* __restrict__ w, _Float16* __restrict__ bt,
                          int K, int N) {
    int e = blockIdx.x * blockDim.x + threadIdx.x;
    if (e >= N * K) return;
    int n = e / K, k = e % K;
    bt[e] = (_Float16)w[k * N + n];
}

// conv weight prep: w [O][I][27] f32 -> wt [27][O][I] f16
__global__ void k_wprep_conv(const float* __restrict__ w, _Float16* __restrict__ wt,
                             int O, int I) {
    int e = blockIdx.x * blockDim.x + threadIdx.x;
    if (e >= 27 * O * I) return;
    int t = e / (O * I);
    int r = e % (O * I);
    int o = r / I, i = r % I;
    wt[e] = (_Float16)w[(o * I + i) * 27 + t];
}

// ---------------------------------------------------------------------------
// GroupNorm stats + affine
// ---------------------------------------------------------------------------
__global__ void k_gn_stats_f32(const float* __restrict__ x, float* __restrict__ stats,
                               int gsz) {
    int bg = blockIdx.x >> 8;
    int ch = blockIdx.x & 255;
    size_t base = (size_t)bg * gsz * RRR;   // group channels contiguous in NCDHW
    float s = 0.f, q = 0.f;
    int e0 = ch * 4096;
    for (int i = threadIdx.x; i < 4096; i += 256) {
        float v = x[base + e0 + i];
        s += v; q += v * v;
    }
    __shared__ float rs[256], rq[256];
    rs[threadIdx.x] = s; rq[threadIdx.x] = q;
    __syncthreads();
    for (int o = 128; o > 0; o >>= 1) {
        if (threadIdx.x < o) { rs[threadIdx.x] += rs[threadIdx.x + o];
                               rq[threadIdx.x] += rq[threadIdx.x + o]; }
        __syncthreads();
    }
    if (threadIdx.x == 0) {
        atomicAdd(&stats[bg * 2 + 0], rs[0]);
        atomicAdd(&stats[bg * 2 + 1], rq[0]);
    }
}

// stats over f16 NDHWC tensor [B][RRR][C], groups of gsz channels
__global__ void k_gn_stats_f16(const _Float16* __restrict__ h, float* __restrict__ stats,
                               int C, int gsz) {
    int bg = blockIdx.x >> 8;
    int ch = blockIdx.x & 255;
    int G = C / gsz;
    int b = bg / G, g = bg % G;
    float s = 0.f, q = 0.f;
    int e0 = ch * 4096;
    for (int i = threadIdx.x; i < 4096; i += 256) {
        int e = e0 + i;                  // e in [0, gsz*RRR)
        int vox = e >> 2;                // gsz==4
        int ci  = e & 3;
        float v = (float)h[((size_t)b * RRR + vox) * C + g * gsz + ci];
        s += v; q += v * v;
    }
    __shared__ float rs[256], rq[256];
    rs[threadIdx.x] = s; rq[threadIdx.x] = q;
    __syncthreads();
    for (int o = 128; o > 0; o >>= 1) {
        if (threadIdx.x < o) { rs[threadIdx.x] += rs[threadIdx.x + o];
                               rq[threadIdx.x] += rq[threadIdx.x + o]; }
        __syncthreads();
    }
    if (threadIdx.x == 0) {
        atomicAdd(&stats[bg * 2 + 0], rs[0]);
        atomicAdd(&stats[bg * 2 + 1], rq[0]);
    }
}

// aff[(b*C+c)*2] = scale, [..+1] = shift
__global__ void k_gn_aff(const float* __restrict__ stats, const float* __restrict__ sc,
                         const float* __restrict__ bi, float* __restrict__ aff,
                         int C, int gsz) {
    int t = threadIdx.x;
    if (t >= BB * C) return;
    int b = t / C, c = t % C;
    int bg = b * (C / gsz) + c / gsz;
    float n  = (float)gsz * (float)RRR;
    float mu = stats[bg * 2] / n;
    float va = stats[bg * 2 + 1] / n - mu * mu;
    float rs = rsqrtf(va + 1e-5f);
    float a  = sc[c] * rs;
    aff[(b * C + c) * 2 + 0] = a;
    aff[(b * C + c) * 2 + 1] = bi[c] - mu * a;
}

// f32 NCDHW -> f16 NDHWC, optional per-(b,c) affine
__global__ void k_ncdhw_to_ndhwc(const float* __restrict__ in,
                                 const float* __restrict__ aff,
                                 _Float16* __restrict__ out, int C) {
    size_t e = (size_t)blockIdx.x * blockDim.x + threadIdx.x;
    if (e >= (size_t)BB * RRR * C) return;
    size_t vox = e / C;
    int c = (int)(e % C);
    int b = (int)(vox >> 18);
    size_t vl = vox & (RRR - 1);
    float v = in[((size_t)b * C + c) * RRR + vl];
    if (aff) v = v * aff[(b * C + c) * 2] + aff[(b * C + c) * 2 + 1];
    out[e] = (_Float16)v;
}

// ---------------------------------------------------------------------------
// 3x3x3 conv, implicit GEMM via WMMA.
// in: f16 NDHWC [B][R][R][R][CIN], wt: f16 [27][COUT][CIN] (B^T per tap)
// EPI 0: out_f16 = relu(acc+bias)                       (conv1)
// EPI 1: v = relu(acc+bias)+res;  out_f32 (NCDHW) + out_f16 (NDHWC)  (conv2)
// EPI 2: v = acc+bias+xres(NCDHW); out_f32 (NCDHW)      (fus)
// ---------------------------------------------------------------------------
template<int CIN, int COUT, int EPI>
__global__ __launch_bounds__(32 * (COUT / 16))
void k_conv3(const _Float16* __restrict__ in, const float* __restrict__ aff,
             const _Float16* __restrict__ wt, const float* __restrict__ bias,
             const float* __restrict__ res, float* __restrict__ out_f32,
             _Float16* __restrict__ out_f16, const float* __restrict__ xres) {
    const int tid = threadIdx.x;
    const int wv  = tid >> 5;
    const int l   = tid & 31;

    int bid = blockIdx.x;
    const int xt = bid & 3;          bid >>= 2;
    const int y  = bid & 63;         bid >>= 6;
    const int z  = bid & 63;
    const int b  = bid >> 6;
    const int x0 = xt * 16;

    __shared__ __align__(16) _Float16 As[16][CIN];
    __shared__ float affs[CIN][2];

    if (aff) {
        for (int c = tid; c < CIN; c += blockDim.x) {
            affs[c][0] = aff[(b * CIN + c) * 2 + 0];
            affs[c][1] = aff[(b * CIN + c) * 2 + 1];
        }
    }

    v8f acc = {};
    for (int tap = 0; tap < 27; ++tap) {
        const int dz = tap / 9 - 1;
        const int dy = (tap / 3) % 3 - 1;
        const int dx = tap % 3 - 1;
        const int zz = z + dz, yy = y + dy;
        const bool rok = ((unsigned)zz < 64u) && ((unsigned)yy < 64u);
        __syncthreads();
        for (int e = tid; e < 16 * CIN; e += blockDim.x) {
            const int m = e / CIN, c = e % CIN;
            const int xx = x0 + m + dx;
            _Float16 v = (_Float16)0.f;
            if (rok && (unsigned)xx < 64u) {
                v = in[(((((size_t)b * 64 + zz) * 64 + yy) * 64 + xx) * CIN) + c];
                if (aff) v = (_Float16)((float)v * affs[c][0] + affs[c][1]);
            }
            As[m][c] = v;
        }
        __syncthreads();
        const _Float16* wb = wt + ((size_t)tap * COUT + wv * 16) * CIN;
        if (tap < 26)
            __builtin_prefetch(wt + ((size_t)(tap + 1) * COUT + wv * 16) * CIN, 0, 0);
#pragma unroll
        for (int kt = 0; kt < CIN / 32; ++kt) {
            v16h a  = frag_a(&As[0][kt * 32], CIN);
            v16h bf = frag_b(wb + kt * 32, CIN);
            acc = wmma16(a, bf, acc);
        }
    }

    const int n0 = wv * 16 + (l & 15);
    const int mb = 8 * (l >> 4);
    const float bn = bias[n0];
#pragma unroll
    for (int i = 0; i < 8; ++i) {
        const int m = mb + i;
        const size_t vox_l = ((size_t)z * 64 + y) * 64 + (x0 + m);
        const size_t vox_g = (size_t)b * RRR + vox_l;
        float v = acc[i] + bn;
        if constexpr (EPI == 0) {
            v = fmaxf(v, 0.f);
            out_f16[vox_g * COUT + n0] = (_Float16)v;
        } else if constexpr (EPI == 1) {
            v = fmaxf(v, 0.f) + res[vox_g * COUT + n0];
            out_f32[((size_t)b * COUT + n0) * RRR + vox_l] = v;
            out_f16[vox_g * COUT + n0] = (_Float16)v;
        } else {
            v += xres[((size_t)b * COUT + n0) * RRR + vox_l];
            out_f32[((size_t)b * COUT + n0) * RRR + vox_l] = v;
        }
    }
}

// ---------------------------------------------------------------------------
// 1x1x1 residual conv: res[vox][64] = xa[vox][32] @ W^T + b
// ---------------------------------------------------------------------------
__global__ __launch_bounds__(128)
void k_res_gemm(const _Float16* __restrict__ xa, const _Float16* __restrict__ wt,
                const float* __restrict__ bias, float* __restrict__ out) {
    const int wv = threadIdx.x >> 5;
    const int l  = threadIdx.x & 31;
    const size_t m0 = (size_t)blockIdx.x * 64 + wv * 16;
    const v16h a = frag_a(xa + m0 * 32, 32);
    const int mb = 8 * (l >> 4);
#pragma unroll
    for (int nt = 0; nt < 4; ++nt) {
        v8f z = {};
        v8f o = wmma16(a, frag_b(wt + nt * 16 * 32, 32), z);
        const int n = nt * 16 + (l & 15);
        const float bn = bias[n];
#pragma unroll
        for (int i = 0; i < 8; ++i)
            out[(m0 + mb + i) * 64 + n] = o[i] + bn;
    }
}

// ---------------------------------------------------------------------------
// Trilinear voxel->point sampling. vox f16 NDHWC [B][RRR][64] -> c0 f32 [B*N][64]
// ---------------------------------------------------------------------------
__global__ void k_sample(const _Float16* __restrict__ vox, const float* __restrict__ xyz,
                         float* __restrict__ c0) {
    const size_t p = (size_t)blockIdx.x * blockDim.x + threadIdx.x;
    if (p >= NPTS) return;
    const int b = (int)(p / NP);
    float pc[3], fr[3];
    int i0[3];
#pragma unroll
    for (int d = 0; d < 3; ++d) {
        float nor = xyz[p * 3 + d] * INV_NORM + 0.5f;
        nor = fminf(fmaxf(nor, 0.f), 1.f);
        float c = nor * 63.f;
        int v0 = (int)floorf(c);
        v0 = min(max(v0, 0), 62);
        pc[d] = c; i0[d] = v0; fr[d] = c - (float)v0;
    }
    float o[64];
#pragma unroll
    for (int c = 0; c < 64; ++c) o[c] = 0.f;
    for (int dz = 0; dz < 2; ++dz)
        for (int dy = 0; dy < 2; ++dy)
            for (int dx = 0; dx < 2; ++dx) {
                const float w = (dx ? fr[0] : 1.f - fr[0]) *
                                (dy ? fr[1] : 1.f - fr[1]) *
                                (dz ? fr[2] : 1.f - fr[2]);
                const size_t vidx = (size_t)b * RRR +
                    (((size_t)(i0[2] + dz) * 64 + (i0[1] + dy)) * 64 + (i0[0] + dx));
                const _Float16* r = vox + vidx * 64;
#pragma unroll
                for (int c = 0; c < 64; ++c) o[c] += w * (float)r[c];
            }
    float* dst = c0 + p * 64;
#pragma unroll
    for (int c = 0; c < 64; ++c) dst[c] = o[c];
}

// ---------------------------------------------------------------------------
// Point MLP: c1 = relu(c0@W1+b1)@W2+b2 + c_res@W3+b3     (WMMA)
// ---------------------------------------------------------------------------
__global__ __launch_bounds__(128)
void k_point_mlp(const float* __restrict__ c0, const float* __restrict__ cres,
                 const _Float16* __restrict__ w1t, const float* __restrict__ b1,
                 const _Float16* __restrict__ w2t, const float* __restrict__ b2,
                 const _Float16* __restrict__ fc2t, const float* __restrict__ b3,
                 float* __restrict__ c1) {
    __shared__ __align__(16) _Float16 A0[4][16][64];
    __shared__ __align__(16) _Float16 H[4][16][128];
    __shared__ __align__(16) _Float16 CR[4][16][32];
    const int wv = threadIdx.x >> 5;
    const int l  = threadIdx.x & 31;
    const size_t r0 = (size_t)blockIdx.x * 64 + wv * 16;

    for (int e = l; e < 16 * 64; e += 32) {
        int m = e >> 6, c = e & 63;
        A0[wv][m][c] = (_Float16)c0[(r0 + m) * 64 + c];
    }
    for (int e = l; e < 16 * 32; e += 32) {
        int m = e >> 5, c = e & 31;
        CR[wv][m][c] = (_Float16)cres[(r0 + m) * 32 + c];
    }
    __syncthreads();

    const int mb = 8 * (l >> 4);
#pragma unroll
    for (int nt = 0; nt < 8; ++nt) {
        v8f acc = {};
#pragma unroll
        for (int kt = 0; kt < 2; ++kt)
            acc = wmma16(frag_a(&A0[wv][0][kt * 32], 64),
                         frag_b(w1t + nt * 16 * 64 + kt * 32, 64), acc);
        const int n = nt * 16 + (l & 15);
        const float bn = b1[n];
#pragma unroll
        for (int i = 0; i < 8; ++i)
            H[wv][mb + i][n] = (_Float16)fmaxf(acc[i] + bn, 0.f);
    }
    __syncthreads();

#pragma unroll
    for (int nt = 0; nt < 4; ++nt) {
        v8f acc = {};
#pragma unroll
        for (int kt = 0; kt < 4; ++kt)
            acc = wmma16(frag_a(&H[wv][0][kt * 32], 128),
                         frag_b(w2t + nt * 16 * 128 + kt * 32, 128), acc);
        acc = wmma16(frag_a(&CR[wv][0][0], 32), frag_b(fc2t + nt * 16 * 32, 32), acc);
        const int n = nt * 16 + (l & 15);
        const float bn = b2[n] + b3[n];
#pragma unroll
        for (int i = 0; i < 8; ++i)
            c1[(r0 + mb + i) * 64 + n] = acc[i] + bn;
    }
}

// ---------------------------------------------------------------------------
// DPT window transformer block: one workgroup per 125-point window (padded 128)
// ---------------------------------------------------------------------------
__global__ __launch_bounds__(256)
void k_dpt(const float* __restrict__ cin, float* __restrict__ cout,
           const int* __restrict__ idx,
           const float* __restrict__ ln1s, const float* __restrict__ ln1b,
           const _Float16* __restrict__ qkvt, const float* __restrict__ qkvb,
           const _Float16* __restrict__ projt, const float* __restrict__ projb,
           const float* __restrict__ ln2s, const float* __restrict__ ln2b,
           const _Float16* __restrict__ m1t, const float* __restrict__ m1b,
           const _Float16* __restrict__ m2t, const float* __restrict__ m2b) {
    extern __shared__ char smem[];
    float*    Wb = (float*)smem;                         // [128][64]  32KB
    _Float16* Hf = (_Float16*)(smem + 32768);            // [128][64]  16KB (LN1 / o)
    _Float16* Qf = Hf + 128 * 64;                        // [128][64]  16KB (q / h2)
    _Float16* Kf = Qf + 128 * 64;                        // [128][64]  16KB
    _Float16* Vt = Kf + 128 * 64;                        // [64][128]  16KB (v^T)
    _Float16* Sf = Vt + 64 * 128;                        // [128][128] 32KB (scores/att/mlp-h)
    int*      ix = (int*)(Sf + 128 * 128);               // [128]

    const int tid = threadIdx.x;
    const int wv  = tid >> 5;
    const int l   = tid & 31;
    const int win = blockIdx.x % (NP / WSZ);
    const int b   = blockIdx.x / (NP / WSZ);

    if (tid < 128)
        ix[tid] = (tid < WSZ) ? idx[b * NP + win * WSZ + tid] : 0;
    __syncthreads();

    for (int e = tid; e < 128 * 64; e += 256) {
        int m = e >> 6;
        Wb[e] = (m < WSZ) ? cin[((size_t)b * NP + ix[m]) * 64 + (e & 63)] : 0.f;
    }
    __syncthreads();

    // LN1 -> Hf
    if (tid < 128) {
        float mu = 0.f;
        for (int c = 0; c < 64; ++c) mu += Wb[tid * 64 + c];
        mu *= (1.f / 64.f);
        float va = 0.f;
        for (int c = 0; c < 64; ++c) { float d = Wb[tid * 64 + c] - mu; va += d * d; }
        va *= (1.f / 64.f);
        const float rs = rsqrtf(va + 1e-5f);
        for (int c = 0; c < 64; ++c)
            Hf[tid * 64 + c] = (_Float16)((Wb[tid * 64 + c] - mu) * rs * ln1s[c] + ln1b[c]);
    }
    __syncthreads();

    const int mb = wv * 16 + 8 * (l >> 4);
    // QKV: 128x192
#pragma unroll
    for (int nt = 0; nt < 12; ++nt) {
        v8f acc = {};
#pragma unroll
        for (int kt = 0; kt < 2; ++kt)
            acc = wmma16(frag_a(Hf + wv * 16 * 64 + kt * 32, 64),
                         frag_b(qkvt + nt * 16 * 64 + kt * 32, 64), acc);
        const int col = nt * 16 + (l & 15);
        const float bn = qkvb[col];
#pragma unroll
        for (int i = 0; i < 8; ++i) {
            const int m = mb + i;
            const float v = acc[i] + bn;
            if (col < 64)       Qf[m * 64 + col] = (_Float16)v;
            else if (col < 128) Kf[m * 64 + (col - 64)] = (_Float16)v;
            else                Vt[(col - 128) * 128 + m] = (_Float16)v;
        }
    }
    __syncthreads();

    const float scale = 0.17677669529663687f;   // 1/sqrt(32)
    for (int hh = 0; hh < 2; ++hh) {
        // scores = q_h @ k_h^T
#pragma unroll
        for (int nt = 0; nt < 8; ++nt) {
            v8f acc = {};
            acc = wmma16(frag_a(Qf + wv * 16 * 64 + hh * 32, 64),
                         frag_b(Kf + nt * 16 * 64 + hh * 32, 64), acc);
            const int n = nt * 16 + (l & 15);
#pragma unroll
            for (int i = 0; i < 8; ++i)
                Sf[(mb + i) * 128 + n] =
                    (_Float16)((n < WSZ) ? acc[i] * scale : -3.0e4f);
        }
        __syncthreads();
        // softmax per row (in place, f16 storage)
        if (tid < 128) {
            float mx = -1e30f;
            for (int n = 0; n < WSZ; ++n) mx = fmaxf(mx, (float)Sf[tid * 128 + n]);
            float sum = 0.f;
            for (int n = 0; n < WSZ; ++n) sum += __expf((float)Sf[tid * 128 + n] - mx);
            const float inv = 1.f / sum;
            for (int n = 0; n < 128; ++n) {
                float v = (n < WSZ) ? __expf((float)Sf[tid * 128 + n] - mx) * inv : 0.f;
                Sf[tid * 128 + n] = (_Float16)v;
            }
        }
        __syncthreads();
        // o_h = att @ v_h  -> Hf columns [32h..32h+31]
#pragma unroll
        for (int nt = 0; nt < 2; ++nt) {
            v8f acc = {};
#pragma unroll
            for (int kt = 0; kt < 4; ++kt)
                acc = wmma16(frag_a(Sf + wv * 16 * 128 + kt * 32, 128),
                             frag_b(Vt + (hh * 32 + nt * 16) * 128 + kt * 32, 128), acc);
            const int n = hh * 32 + nt * 16 + (l & 15);
#pragma unroll
            for (int i = 0; i < 8; ++i)
                Hf[(mb + i) * 64 + n] = (_Float16)acc[i];
        }
        __syncthreads();
    }

    // proj: Wb += o @ P + pb
#pragma unroll
    for (int nt = 0; nt < 4; ++nt) {
        v8f acc = {};
#pragma unroll
        for (int kt = 0; kt < 2; ++kt)
            acc = wmma16(frag_a(Hf + wv * 16 * 64 + kt * 32, 64),
                         frag_b(projt + nt * 16 * 64 + kt * 32, 64), acc);
        const int n = nt * 16 + (l & 15);
        const float bn = projb[n];
#pragma unroll
        for (int i = 0; i < 8; ++i)
            Wb[(mb + i) * 64 + n] += acc[i] + bn;
    }
    __syncthreads();

    // LN2 -> Qf
    if (tid < 128) {
        float mu = 0.f;
        for (int c = 0; c < 64; ++c) mu += Wb[tid * 64 + c];
        mu *= (1.f / 64.f);
        float va = 0.f;
        for (int c = 0; c < 64; ++c) { float d = Wb[tid * 64 + c] - mu; va += d * d; }
        va *= (1.f / 64.f);
        const float rs = rsqrtf(va + 1e-5f);
        for (int c = 0; c < 64; ++c)
            Qf[tid * 64 + c] = (_Float16)((Wb[tid * 64 + c] - mu) * rs * ln2s[c] + ln2b[c]);
    }
    __syncthreads();

    // MLP1 (relu) -> Sf [128][128]
#pragma unroll
    for (int nt = 0; nt < 8; ++nt) {
        v8f acc = {};
#pragma unroll
        for (int kt = 0; kt < 2; ++kt)
            acc = wmma16(frag_a(Qf + wv * 16 * 64 + kt * 32, 64),
                         frag_b(m1t + nt * 16 * 64 + kt * 32, 64), acc);
        const int n = nt * 16 + (l & 15);
        const float bn = m1b[n];
#pragma unroll
        for (int i = 0; i < 8; ++i)
            Sf[(mb + i) * 128 + n] = (_Float16)fmaxf(acc[i] + bn, 0.f);
    }
    __syncthreads();

    // MLP2 + residual, scatter out
#pragma unroll
    for (int nt = 0; nt < 4; ++nt) {
        v8f acc = {};
#pragma unroll
        for (int kt = 0; kt < 4; ++kt)
            acc = wmma16(frag_a(Sf + wv * 16 * 128 + kt * 32, 128),
                         frag_b(m2t + nt * 16 * 128 + kt * 32, 128), acc);
        const int n = nt * 16 + (l & 15);
        const float bn = m2b[n];
#pragma unroll
        for (int i = 0; i < 8; ++i) {
            const int m = mb + i;
            if (m < WSZ)
                cout[((size_t)b * NP + ix[m]) * 64 + n] = Wb[m * 64 + n] + acc[i] + bn;
        }
    }
}

// ---------------------------------------------------------------------------
// feature_to_voxel scatter-mean + finalize
// ---------------------------------------------------------------------------
__global__ void k_scatter_add(const float* __restrict__ c, const float* __restrict__ xyz,
                              float* __restrict__ acc, float* __restrict__ cnt) {
    const size_t p = (size_t)blockIdx.x * blockDim.x + threadIdx.x;
    if (p >= NPTS) return;
    const int b = (int)(p / NP);
    int v[3];
#pragma unroll
    for (int d = 0; d < 3; ++d) {
        float nor = xyz[p * 3 + d] * INV_NORM + 0.5f;
        int vi = (int)floorf(nor * 64.f);
        v[d] = min(max(vi, 0), 63);
    }
    const size_t flat = ((size_t)v[2] * 64 + v[1]) * 64 + v[0];
    atomicAdd(&cnt[(size_t)b * RRR + flat], 1.f);
    float* dst = acc + ((size_t)b * RRR + flat) * 64;
    const float* src = c + p * 64;
#pragma unroll
    for (int ch = 0; ch < 64; ++ch) atomicAdd(&dst[ch], src[ch]);
}

__global__ void k_voxel_mean(const float* __restrict__ acc, const float* __restrict__ cnt,
                             _Float16* __restrict__ out) {
    const size_t e = (size_t)blockIdx.x * blockDim.x + threadIdx.x;
    if (e >= (size_t)BB * RRR * 64) return;
    const size_t vox = e >> 6;
    out[e] = (_Float16)(acc[e] / fmaxf(cnt[vox], 1.f));
}

// ---------------------------------------------------------------------------
// 2x2x2 maxpool, NCDHW
// ---------------------------------------------------------------------------
__global__ void k_maxpool(const float* __restrict__ in, float* __restrict__ out) {
    const size_t e = (size_t)blockIdx.x * blockDim.x + threadIdx.x;
    if (e >= (size_t)BB * 64 * 32 * 32 * 32) return;
    const int x = (int)(e & 31);
    const int y = (int)((e >> 5) & 31);
    const int z = (int)((e >> 10) & 31);
    const int c = (int)((e >> 15) & 63);
    const int b = (int)(e >> 21);
    const size_t base = (((size_t)b * 64 + c) * 64 + 2 * z) * 4096 +
                        (size_t)(2 * y) * 64 + 2 * x;
    float m = -3.4e38f;
#pragma unroll
    for (int dz = 0; dz < 2; ++dz)
#pragma unroll
        for (int dy = 0; dy < 2; ++dy)
#pragma unroll
            for (int dx = 0; dx < 2; ++dx)
                m = fmaxf(m, in[base + (size_t)dz * 4096 + dy * 64 + dx]);
    out[e] = m;
}

// ---------------------------------------------------------------------------
// Host launcher
// ---------------------------------------------------------------------------
extern "C" void kernel_launch(void* const* d_in, const int* in_sizes, int n_in,
                              void* d_out, int out_size, void* d_ws, size_t ws_size,
                              hipStream_t stream) {
    (void)in_sizes; (void)n_in; (void)out_size; (void)ws_size;

    const float* x        = (const float*)d_in[0];
    const float* c_res    = (const float*)d_in[1];
    const float* xyz      = (const float*)d_in[2];
    const float* x_after  = (const float*)d_in[3];
    const int*   indices  = (const int*)d_in[4];
    const float* gn1_s = (const float*)d_in[6],  * gn1_b = (const float*)d_in[7];
    const float* conv1_w = (const float*)d_in[8],* conv1_b = (const float*)d_in[9];
    const float* gn2_s = (const float*)d_in[10], * gn2_b = (const float*)d_in[11];
    const float* conv2_w = (const float*)d_in[12],*conv2_b = (const float*)d_in[13];
    const float* res_w = (const float*)d_in[14], * res_b = (const float*)d_in[15];
    const float* fc1_w1 = (const float*)d_in[16],* fc1_b1 = (const float*)d_in[17];
    const float* fc1_w2 = (const float*)d_in[18],* fc1_b2 = (const float*)d_in[19];
    const float* fc2_w = (const float*)d_in[20], * fc2_b = (const float*)d_in[21];
    const float* fus_w = (const float*)d_in[22], * fus_b = (const float*)d_in[23];
    const float* ln1_s = (const float*)d_in[24], * ln1_b = (const float*)d_in[25];
    const float* qkv_w = (const float*)d_in[26], * qkv_b = (const float*)d_in[27];
    const float* proj_w = (const float*)d_in[28],* proj_b = (const float*)d_in[29];
    const float* ln2_s = (const float*)d_in[30], * ln2_b = (const float*)d_in[31];
    const float* mlp_w1 = (const float*)d_in[32],* mlp_b1 = (const float*)d_in[33];
    const float* mlp_w2 = (const float*)d_in[34],* mlp_b2 = (const float*)d_in[35];

    // ---- d_out carving: (x_pooled, x_before_pool, x_after, c) f32 ----
    float* o_pool   = (float*)d_out;
    float* o_xbp    = o_pool + (size_t)BB * 64 * 32768;
    float* o_xafter = o_xbp + (size_t)BB * 64 * RRR;
    float* o_c      = o_xafter + (size_t)BB * 64 * RRR;

    // ---- workspace carving ----
    char* w = (char*)d_ws;
    auto alloc = [&](size_t bytes) -> char* {
        char* p = w;
        w += (bytes + 255) & ~(size_t)255;
        return p;
    };
    float*    stats  = (float*)alloc(64 * 4);           // [16][2] gn1, then gn2
    float*    stats2 = stats + 32;
    float*    aff1   = (float*)alloc(BB * 32 * 2 * 4);
    float*    aff2   = (float*)alloc(BB * 32 * 2 * 4);
    _Float16* wt1    = (_Float16*)alloc(27 * 32 * 32 * 2);
    _Float16* wt2    = (_Float16*)alloc(27 * 64 * 32 * 2);
    _Float16* wtf    = (_Float16*)alloc(27 * 64 * 64 * 2);
    _Float16* reswt  = (_Float16*)alloc(64 * 32 * 2);
    _Float16* w1t    = (_Float16*)alloc(128 * 64 * 2);
    _Float16* w2t    = (_Float16*)alloc(64 * 128 * 2);
    _Float16* fc2t   = (_Float16*)alloc(64 * 32 * 2);
    _Float16* qkvt   = (_Float16*)alloc(3 * 192 * 64 * 2);
    _Float16* projt  = (_Float16*)alloc(3 * 64 * 64 * 2);
    _Float16* m1t    = (_Float16*)alloc(3 * 128 * 64 * 2);
    _Float16* m2t    = (_Float16*)alloc(3 * 64 * 128 * 2);
    _Float16* t_x1   = (_Float16*)alloc((size_t)BRRR * 32 * 2);
    _Float16* t_xa   = (_Float16*)alloc((size_t)BRRR * 32 * 2);
    _Float16* t_h1   = (_Float16*)alloc((size_t)BRRR * 32 * 2);
    float*    resbuf = (float*)alloc((size_t)BRRR * 64 * 4);
    _Float16* t_x    = (_Float16*)alloc((size_t)BRRR * 64 * 2);
    float*    c0     = (float*)alloc((size_t)NPTS * 64 * 4);
    float*    c1     = (float*)alloc((size_t)NPTS * 64 * 4);
    float*    c2     = (float*)alloc((size_t)NPTS * 64 * 4);
    float*    vacc   = (float*)alloc((size_t)BRRR * 64 * 4);
    float*    vcnt   = (float*)alloc((size_t)BRRR * 4);
    _Float16* t_fus  = (_Float16*)alloc((size_t)BRRR * 64 * 2);

    // ---- zero accumulators ----
    k_zero_f32<<<1, 64, 0, stream>>>(stats, 64);
    k_zero_f32<<<32768, 256, 0, stream>>>(vacc, (size_t)BRRR * 64 + BRRR); // vacc+vcnt contiguous

    // ---- weight prep ----
    k_wprep_conv<<<(27 * 32 * 32 + 255) / 256, 256, 0, stream>>>(conv1_w, wt1, 32, 32);
    k_wprep_conv<<<(27 * 64 * 32 + 255) / 256, 256, 0, stream>>>(conv2_w, wt2, 64, 32);
    k_wprep_conv<<<(27 * 64 * 64 + 255) / 256, 256, 0, stream>>>(fus_w, wtf, 64, 64);
    k_cvt16<<<(64 * 32 + 255) / 256, 256, 0, stream>>>(res_w, reswt, 64 * 32);
    k_wprep_t<<<(128 * 64 + 255) / 256, 256, 0, stream>>>(fc1_w1, w1t, 64, 128);
    k_wprep_t<<<(64 * 128 + 255) / 256, 256, 0, stream>>>(fc1_w2, w2t, 128, 64);
    k_wprep_t<<<(64 * 32 + 255) / 256, 256, 0, stream>>>(fc2_w, fc2t, 32, 64);
    for (int i = 0; i < 3; ++i) {
        k_wprep_t<<<(192 * 64 + 255) / 256, 256, 0, stream>>>(qkv_w + i * 64 * 192,
                                                              qkvt + i * 192 * 64, 64, 192);
        k_wprep_t<<<(64 * 64 + 255) / 256, 256, 0, stream>>>(proj_w + i * 64 * 64,
                                                             projt + i * 64 * 64, 64, 64);
        k_wprep_t<<<(128 * 64 + 255) / 256, 256, 0, stream>>>(mlp_w1 + i * 64 * 128,
                                                              m1t + i * 128 * 64, 64, 128);
        k_wprep_t<<<(64 * 128 + 255) / 256, 256, 0, stream>>>(mlp_w2 + i * 128 * 64,
                                                              m2t + i * 64 * 128, 128, 64);
    }

    // ---- GN1 + layout transform ----
    k_gn_stats_f32<<<16 * 256, 256, 0, stream>>>(x, stats, 4);
    k_gn_aff<<<1, 64, 0, stream>>>(stats, gn1_s, gn1_b, aff1, 32, 4);
    k_ncdhw_to_ndhwc<<<(int)(((size_t)BRRR * 32 + 255) / 256), 256, 0, stream>>>(x, aff1, t_x1, 32);
    k_ncdhw_to_ndhwc<<<(int)(((size_t)BRRR * 32 + 255) / 256), 256, 0, stream>>>(x_after, nullptr, t_xa, 32);

    // ---- conv1 (gn1 folded in input) ----
    k_conv3<32, 32, 0><<<32768, 64, 0, stream>>>(t_x1, nullptr, wt1, conv1_b,
                                                 nullptr, nullptr, t_h1, nullptr);
    // ---- GN2 stats from conv1 output ----
    k_gn_stats_f16<<<16 * 256, 256, 0, stream>>>(t_h1, stats2, 32, 4);
    k_gn_aff<<<1, 64, 0, stream>>>(stats2, gn2_s, gn2_b, aff2, 32, 4);

    // ---- residual 1x1 conv ----
    k_res_gemm<<<BRRR / 64, 128, 0, stream>>>(t_xa, reswt, res_b, resbuf);

    // ---- conv2 (gn2 affine fused) + residual -> x_after ----
    k_conv3<32, 64, 1><<<32768, 128, 0, stream>>>(t_h1, aff2, wt2, conv2_b,
                                                  resbuf, o_xafter, t_x, nullptr);

    // ---- voxel->point sample + point MLP ----
    k_sample<<<(NPTS + 255) / 256, 256, 0, stream>>>(t_x, xyz, c0);
    k_point_mlp<<<NPTS / 64, 128, 0, stream>>>(c0, c_res, w1t, fc1_b1, w2t, fc1_b2,
                                               fc2t, fc2_b, c1);

    // ---- 3 DPT blocks ----
    const size_t dpt_smem = 32768 + 4 * (128 * 64 * 2) + 128 * 128 * 2 + 512;
    const float* dpt_in[3]  = {c1, c2, c1};
    float*       dpt_out[3] = {c2, c1, o_c};
    for (int i = 0; i < 3; ++i) {
        k_dpt<<<BB * (NP / WSZ), 256, dpt_smem, stream>>>(
            dpt_in[i], dpt_out[i], indices + i * NPTS,
            ln1_s + i * 64, ln1_b + i * 64,
            qkvt + i * 192 * 64, qkv_b + i * 192,
            projt + i * 64 * 64, proj_b + i * 64,
            ln2_s + i * 64, ln2_b + i * 64,
            m1t + i * 128 * 64, mlp_b1 + i * 128,
            m2t + i * 64 * 128, mlp_b2 + i * 64);
    }

    // ---- feature_to_voxel scatter-mean ----
    k_scatter_add<<<(NPTS + 255) / 256, 256, 0, stream>>>(o_c, xyz, vacc, vcnt);
    k_voxel_mean<<<(int)(((size_t)BRRR * 64 + 255) / 256), 256, 0, stream>>>(vacc, vcnt, t_fus);

    // ---- fusion conv + residual -> x_before_pool ----
    k_conv3<64, 64, 2><<<32768, 128, 0, stream>>>(t_fus, nullptr, wtf, fus_b,
                                                  nullptr, o_xbp, nullptr, o_xafter);

    // ---- maxpool -> x ----
    k_maxpool<<<(int)(((size_t)BB * 64 * 32768 + 255) / 256), 256, 0, stream>>>(o_xbp, o_pool);
}
